// CoreTokenAttention_test_6975026889300
// MI455X (gfx1250) — matrix operand order repaired and checked by reading, not compiled, not run on hardware
//
#include <hip/hip_runtime.h>

// Problem constants (from reference)
constexpr int CB = 2, CS = 2048, CD = 1024, CH = 16, CHD = 64, CR = 1024;
constexpr int CBH = CB * CH;        // 32 (b,h) pairs
constexpr int CRNEW = 921;          // int(1024 * round(0.5**(1/6), 1)) = int(921.6)

typedef __bf16 bf16_t;
typedef __bf16 v16bf  __attribute__((ext_vector_type(16)));
typedef __bf16 bf16x8 __attribute__((ext_vector_type(8)));
typedef float  v8f    __attribute__((ext_vector_type(8)));

// ---------- optional gfx1250 async global->LDS path (guarded, with fallback) ----------
#if defined(__HIP_DEVICE_COMPILE__) && __has_builtin(__builtin_amdgcn_global_load_async_to_lds_b128)
#define HAVE_ASYNC_LDS 1
// Builtin signature (from hipcc diagnostic): (v4i addrspace(1)*, v4i addrspace(3)*, imm, imm)
typedef int v4i_gcc __attribute__((vector_size(16)));
typedef __attribute__((address_space(1))) v4i_gcc* as1_v4ip;
typedef __attribute__((address_space(3))) v4i_gcc* as3_v4ip;
static __device__ inline void async_copy16(const bf16_t* g, bf16_t* l) {
  // 16B DMA global -> LDS, bypasses VGPRs, tracked by ASYNCcnt.
  // AS1 pointer: same 64-bit value as generic. AS3 pointer: low 32 bits of the
  // generic LDS address are the LDS byte offset (aperture encoding).
  __builtin_amdgcn_global_load_async_to_lds_b128(
      (as1_v4ip)(uintptr_t)g, (as3_v4ip)(unsigned int)(uintptr_t)l, 0, 0);
}
static __device__ inline void wait_async(int n) {
#if __has_builtin(__builtin_amdgcn_s_wait_asynccnt)
  if (n == 0) __builtin_amdgcn_s_wait_asynccnt(0);
  else        __builtin_amdgcn_s_wait_asynccnt(1);
#else
  if (n == 0) asm volatile("s_wait_asynccnt 0x0" ::: "memory");
  else        asm volatile("s_wait_asynccnt 0x1" ::: "memory");
#endif
}
#else
#define HAVE_ASYNC_LDS 0
#endif

static __device__ inline v16bf frag_cat(bf16x8 lo, bf16x8 hi) {
  return __builtin_shufflevector(lo, hi, 0,1,2,3,4,5,6,7,8,9,10,11,12,13,14,15);
}

// A fragment (16x32 MxK, bf16, wave32): lane&15 = row; lanes>=16 take K-halves 8..15/24..31
static __device__ inline v16bf load_afrag(const bf16_t* base, int ld, int row0, int k0, int lane) {
  const bf16_t* p = base + (size_t)(row0 + (lane & 15)) * ld + k0 + ((lane >> 4) * 8);
  bf16x8 lo = *reinterpret_cast<const bf16x8*>(p);
  bf16x8 hi = *reinterpret_cast<const bf16x8*>(p + 16);
  return frag_cat(lo, hi);
}

// B fragment (32x16 KxN) sourced from row-major Bmat[N][ld] (i.e. we compute A·Bmatᵀ):
// lane&15 = column N; lanes>=16 take K-halves 8..15/24..31 (same K order as A)
static __device__ inline v16bf load_bfrag(const bf16_t* base, int ld, int col0, int k0, int lane) {
  const bf16_t* p = base + (size_t)(col0 + (lane & 15)) * ld + k0 + ((lane >> 4) * 8);
  bf16x8 lo = *reinterpret_cast<const bf16x8*>(p);
  bf16x8 hi = *reinterpret_cast<const bf16x8*>(p + 16);
  return frag_cat(lo, hi);
}

static __device__ inline v8f wmma_bf16(v16bf a, v16bf b, v8f c) {
  return __builtin_amdgcn_wmma_f32_16x16x32_bf16(false, a, false, b, (short)0, c, false, false);
}

static __device__ inline void atomicMaxF(float* addr, float v) {
  if (v >= 0.f) atomicMax((int*)addr, __float_as_int(v));
  else          atomicMin((unsigned int*)addr, __float_as_uint(v));
}

// ---------------- f32 -> bf16 conversion ----------------
__global__ void k_cvt_bf16(const float* __restrict__ in, bf16_t* __restrict__ out, int n) {
  int i = blockIdx.x * blockDim.x + threadIdx.x;
  if (i < n) out[i] = (bf16_t)in[i];
}

// ---------------- GEMM: C[M,N] = A[M,K] · W[N,K]ᵀ + bias ----------------
// Block tile 128x64 (8 waves, 32x32 per wave). Weight tile shared by all 8 waves:
// staged via async global->LDS DMA (double-buffered) when available.
// mode 0: bf16 out in head layout [b][h][s][64];  mode 1: fp32 out row-major (final output)
__global__ void k_gemm(const bf16_t* __restrict__ A, const bf16_t* __restrict__ W,
                       const float* __restrict__ bias, bf16_t* __restrict__ outb,
                       float* __restrict__ outf, int M, int N, int K, int mode) {
  const int lane = threadIdx.x & 31;
  const int wave = threadIdx.x >> 5;              // 8 waves
  const int m0   = blockIdx.x * 128 + (wave & 3) * 32;
  const int nblk = blockIdx.y * 64;
  const int n0   = (wave >> 2) * 32;              // column offset inside the block tile
  v8f c00 = {}, c01 = {}, c10 = {}, c11 = {};

#if HAVE_ASYNC_LDS
  // 64 cols x 32 k of bf16 = 4KB per buffer; 256 threads x 16B fill one buffer exactly.
  __shared__ alignas(16) bf16_t wtile[2][64][32];
  const int tn = threadIdx.x >> 2;                // 0..63 : weight row (output column)
  const int tc = threadIdx.x & 3;                 // 0..3  : 16B chunk within 32 k-values
  const bf16_t* wsrc = W + (size_t)(nblk + tn) * K + tc * 8;
  bf16_t* ldst0 = &wtile[0][tn][tc * 8];
  bf16_t* ldst1 = &wtile[1][tn][tc * 8];
  async_copy16(wsrc, ldst0);                      // prologue: tile for k0=0
  for (int k0 = 0; k0 < K; k0 += 32) {
    const int  cur  = (k0 >> 5) & 1;
    const bool more = (k0 + 32) < K;
    if (more) async_copy16(wsrc + k0 + 32, cur ? ldst0 : ldst1);
    wait_async(more ? 1 : 0);                     // current tile complete (in-order)
    __syncthreads();
    v16bf a0 = load_afrag(A, K, m0,      k0, lane);
    v16bf a1 = load_afrag(A, K, m0 + 16, k0, lane);
    const bf16_t* wt = &wtile[cur][0][0];
    v16bf b0 = load_bfrag(wt, 32, n0,      0, lane);
    v16bf b1 = load_bfrag(wt, 32, n0 + 16, 0, lane);
    c00 = wmma_bf16(a0, b0, c00);
    c01 = wmma_bf16(a0, b1, c01);
    c10 = wmma_bf16(a1, b0, c10);
    c11 = wmma_bf16(a1, b1, c11);
    __syncthreads();                              // protect buffer before next overwrite
  }
#else
  for (int k0 = 0; k0 < K; k0 += 32) {
    v16bf a0 = load_afrag(A, K, m0,      k0, lane);
    v16bf a1 = load_afrag(A, K, m0 + 16, k0, lane);
    v16bf b0 = load_bfrag(W, K, nblk + n0,      k0, lane);
    v16bf b1 = load_bfrag(W, K, nblk + n0 + 16, k0, lane);
    c00 = wmma_bf16(a0, b0, c00);
    c01 = wmma_bf16(a0, b1, c01);
    c10 = wmma_bf16(a1, b0, c10);
    c11 = wmma_bf16(a1, b1, c11);
  }
#endif

  const int mrow = (lane >> 4) * 8;
  const int ncol = lane & 15;
  v8f cs[2][2] = {{c00, c01}, {c10, c11}};
#pragma unroll
  for (int i = 0; i < 2; ++i)
#pragma unroll
    for (int j = 0; j < 2; ++j) {
      int col = nblk + n0 + j * 16 + ncol;
      float bv = bias[col];
#pragma unroll
      for (int v = 0; v < 8; ++v) {
        int row = m0 + i * 16 + mrow + v;
        float val = cs[i][j][v] + bv;
        if (mode == 0) {
          int b = row / CS, q = row % CS;
          int h = col >> 6, d = col & 63;
          outb[((size_t)(b * CH + h) * CS + q) * CHD + d] = (bf16_t)val;
        } else {
          outf[(size_t)row * N + col] = val;
        }
      }
    }
}

// ---------------- gather Kp rows, transposed Vp, mask; init stats ----------------
__global__ void k_gather(const bf16_t* __restrict__ Kw, const bf16_t* __restrict__ Vw,
                         const int* __restrict__ topk, const int* __restrict__ mask,
                         bf16_t* __restrict__ Kp, bf16_t* __restrict__ VpT,
                         int* __restrict__ mg, float* __restrict__ colmax) {
  int t = blockIdx.x * blockDim.x + threadIdx.x;
  if (t >= CBH * CR) return;
  int bh = t / CR, r = t % CR;
  int b  = bh / CH;
  int idx = topk[t];
  const bf16_t* ksrc = Kw + ((size_t)bh * CS + idx) * CHD;
  bf16_t* kdst = Kp + (size_t)t * CHD;
#pragma unroll
  for (int d = 0; d < CHD; d += 8)
    *reinterpret_cast<bf16x8*>(kdst + d) = *reinterpret_cast<const bf16x8*>(ksrc + d);
  const bf16_t* vsrc = Vw + ((size_t)bh * CS + idx) * CHD;
  bf16_t* vp = VpT + (size_t)bh * CHD * CR;      // [64][R]
  for (int d = 0; d < CHD; ++d) vp[(size_t)d * CR + r] = vsrc[d];
  mg[t] = mask[b * CS + idx];
  colmax[t] = -__builtin_inff();
}

// ---------------- shared score-tile computation (masked + scaled) ----------------
static __device__ inline void score_tiles(const bf16_t* Kp_bh, const int* mg_bh,
                                          v16bf qa0, v16bf qa1, int r0, int lane,
                                          v8f& s0, v8f& s1) {
  v8f z0 = {}, z1 = {};
  v16bf kb;
  kb = load_bfrag(Kp_bh, CHD, r0,      0,  lane); z0 = wmma_bf16(qa0, kb, z0);
  kb = load_bfrag(Kp_bh, CHD, r0,      32, lane); z0 = wmma_bf16(qa1, kb, z0);
  kb = load_bfrag(Kp_bh, CHD, r0 + 16, 0,  lane); z1 = wmma_bf16(qa0, kb, z1);
  kb = load_bfrag(Kp_bh, CHD, r0 + 16, 32, lane); z1 = wmma_bf16(qa1, kb, z1);
  const float scale = 0.125f;                     // 1/sqrt(64)
  int c0 = r0 + (lane & 15), c1 = c0 + 16;
  bool k0 = mg_bh[c0] != 0, k1 = mg_bh[c1] != 0;
#pragma unroll
  for (int v = 0; v < 8; ++v) {
    s0[v] = k0 ? z0[v] * scale : -1e10f;
    s1[v] = k1 ? z1[v] * scale : -1e10f;
  }
}

// ---------------- pass 1: column max over all queries ----------------
__global__ void k_pass1(const bf16_t* __restrict__ Q, const bf16_t* __restrict__ Kp,
                        const int* __restrict__ mg, float* __restrict__ colmax) {
  const int lane = threadIdx.x & 31, wave = threadIdx.x >> 5;
  const int bh = blockIdx.y;
  const int q0 = blockIdx.x * 64 + wave * 16;
  const bf16_t* Qbh  = Q  + (size_t)bh * CS * CHD;
  const bf16_t* Kpbh = Kp + (size_t)bh * CR * CHD;
  const int* mgbh = mg + bh * CR;
  float* cm = colmax + bh * CR;
  v16bf qa0 = load_afrag(Qbh, CHD, q0, 0,  lane);
  v16bf qa1 = load_afrag(Qbh, CHD, q0, 32, lane);
  for (int r0 = 0; r0 < CR; r0 += 32) {
    if (r0 + 32 < CR) __builtin_prefetch(Kpbh + (size_t)(r0 + 32 + lane) * CHD, 0, 1);
    v8f s0, s1;
    score_tiles(Kpbh, mgbh, qa0, qa1, r0, lane, s0, s1);
    float m0 = s0[0], m1 = s1[0];
#pragma unroll
    for (int v = 1; v < 8; ++v) { m0 = fmaxf(m0, s0[v]); m1 = fmaxf(m1, s1[v]); }
    atomicMaxF(cm + r0 +      (lane & 15), m0);   // fp max is order-independent
    atomicMaxF(cm + r0 + 16 + (lane & 15), m1);
  }
}

// ---------------- pass 2: column exp-sums (deterministic partials per q-tile) ----------------
__global__ void k_pass2(const bf16_t* __restrict__ Q, const bf16_t* __restrict__ Kp,
                        const int* __restrict__ mg, const float* __restrict__ colmax,
                        float* __restrict__ partial /* [32 qtiles][CBH*CR] */) {
  __shared__ float psum[32][8];
  const int lane = threadIdx.x & 31, wave = threadIdx.x >> 5;
  const int bh = blockIdx.y;
  const int q0 = blockIdx.x * 64 + wave * 16;
  const bf16_t* Qbh  = Q  + (size_t)bh * CS * CHD;
  const bf16_t* Kpbh = Kp + (size_t)bh * CR * CHD;
  const int* mgbh = mg + bh * CR;
  const float* cm = colmax + bh * CR;
  v16bf qa0 = load_afrag(Qbh, CHD, q0, 0,  lane);
  v16bf qa1 = load_afrag(Qbh, CHD, q0, 32, lane);
  for (int r0 = 0; r0 < CR; r0 += 32) {
    v8f s0, s1;
    score_tiles(Kpbh, mgbh, qa0, qa1, r0, lane, s0, s1);
    int c0 = r0 + (lane & 15), c1 = c0 + 16;
    float cm0 = cm[c0], cm1 = cm[c1];
    float e0 = 0.f, e1 = 0.f;
#pragma unroll
    for (int v = 0; v < 8; ++v) { e0 += __expf(s0[v] - cm0); e1 += __expf(s1[v] - cm1); }
    psum[(lane & 15)     ][wave * 2 + (lane >> 4)] = e0;
    psum[16 + (lane & 15)][wave * 2 + (lane >> 4)] = e1;
    __syncthreads();
    if (threadIdx.x < 32) {
      float s = 0.f;
#pragma unroll
      for (int t = 0; t < 8; ++t) s += psum[threadIdx.x][t];
      partial[(size_t)blockIdx.x * (CBH * CR) + (size_t)bh * CR + r0 + threadIdx.x] = s;
    }
    __syncthreads();
  }
}

// ---------------- fixed-order reduction over 32 q-tile partials ----------------
__global__ void k_reduce32(const float* __restrict__ partial, float* __restrict__ out, int n) {
  int i = blockIdx.x * blockDim.x + threadIdx.x;
  if (i >= n) return;
  float s = 0.f;
  for (int t = 0; t < 32; ++t) s += partial[(size_t)t * n + i];
  out[i] = s;
}

// ---------------- pass 3: attn = exp/colsum; out += attn·Vp (WMMA); importance partials ----------------
__global__ void k_pass3(const bf16_t* __restrict__ Q, const bf16_t* __restrict__ Kp,
                        const bf16_t* __restrict__ VpT, const int* __restrict__ mg,
                        const float* __restrict__ colmax, const float* __restrict__ colsum,
                        bf16_t* __restrict__ attnout, float* __restrict__ imp_partial) {
  __shared__ alignas(16) bf16_t atile[4][16][32];   // per-wave attn tile (q x r), bf16
  __shared__ float psum[32][8];
  const int lane = threadIdx.x & 31, wave = threadIdx.x >> 5;
  const int bh = blockIdx.y;
  const int q0 = blockIdx.x * 64 + wave * 16;
  const bf16_t* Qbh   = Q   + (size_t)bh * CS * CHD;
  const bf16_t* Kpbh  = Kp  + (size_t)bh * CR * CHD;
  const bf16_t* VpTbh = VpT + (size_t)bh * CHD * CR;
  const int* mgbh = mg + bh * CR;
  const float* cm = colmax + bh * CR;
  const float* cs = colsum + bh * CR;
  v16bf qa0 = load_afrag(Qbh, CHD, q0, 0,  lane);
  v16bf qa1 = load_afrag(Qbh, CHD, q0, 32, lane);
  v8f o0 = {}, o1 = {}, o2 = {}, o3 = {};
  for (int r0 = 0; r0 < CR; r0 += 32) {
    if (r0 + 32 < CR) __builtin_prefetch(Kpbh + (size_t)(r0 + 32 + lane) * CHD, 0, 1);
    v8f s0, s1;
    score_tiles(Kpbh, mgbh, qa0, qa1, r0, lane, s0, s1);
    int c0 = r0 + (lane & 15), c1 = c0 + 16;
    float cm0 = cm[c0], cm1 = cm[c1];
    float inv0 = 1.f / cs[c0], inv1 = 1.f / cs[c1];
    float e0 = 0.f, e1 = 0.f;
#pragma unroll
    for (int v = 0; v < 8; ++v) {
      float a0 = __expf(s0[v] - cm0) * inv0;
      float a1 = __expf(s1[v] - cm1) * inv1;
      e0 += a0; e1 += a1;
      int row = (lane >> 4) * 8 + v;                 // C/D layout -> tile row
      atile[wave][row][(lane & 15)]      = (bf16_t)a0;
      atile[wave][row][16 + (lane & 15)] = (bf16_t)a1;
    }
    psum[(lane & 15)     ][wave * 2 + (lane >> 4)] = e0;
    psum[16 + (lane & 15)][wave * 2 + (lane >> 4)] = e1;
    __syncthreads();
    if (threadIdx.x < 32) {
      float s = 0.f;
#pragma unroll
      for (int t = 0; t < 8; ++t) s += psum[threadIdx.x][t];
      imp_partial[(size_t)blockIdx.x * (CBH * CR) + (size_t)bh * CR + r0 + threadIdx.x] =
          s * (1.f / (float)CS);
    }
    // Re-read attn tile in A-fragment layout (16x32, K = r)
    const bf16_t* p = &atile[wave][lane & 15][(lane >> 4) * 8];
    bf16x8 alo = *reinterpret_cast<const bf16x8*>(p);
    bf16x8 ahi = *reinterpret_cast<const bf16x8*>(p + 16);
    v16bf aa = frag_cat(alo, ahi);
    o0 = wmma_bf16(aa, load_bfrag(VpTbh, CR, 0,  r0, lane), o0);
    o1 = wmma_bf16(aa, load_bfrag(VpTbh, CR, 16, r0, lane), o1);
    o2 = wmma_bf16(aa, load_bfrag(VpTbh, CR, 32, r0, lane), o2);
    o3 = wmma_bf16(aa, load_bfrag(VpTbh, CR, 48, r0, lane), o3);
    __syncthreads();
  }
  int b = bh / CH, h = bh % CH;
  v8f oo[4] = {o0, o1, o2, o3};
#pragma unroll
  for (int j = 0; j < 4; ++j) {
    int d = j * 16 + (lane & 15);
#pragma unroll
    for (int v = 0; v < 8; ++v) {
      int q = q0 + (lane >> 4) * 8 + v;
      attnout[((size_t)(b * CS + q)) * CD + h * CHD + d] = (bf16_t)oo[j][v];
    }
  }
}

// ---------------- exact top-k (k=921) by stable rank counting ----------------
__global__ void k_topk(const float* __restrict__ importance, int* __restrict__ outIdx) {
  __shared__ float v[CR];
  int bh = blockIdx.x;
  const float* imp = importance + (size_t)bh * CR;
  for (int i = threadIdx.x; i < CR; i += blockDim.x) v[i] = imp[i];
  __syncthreads();
  for (int i = threadIdx.x; i < CR; i += blockDim.x) {
    float vi = v[i];
    int rank = 0;
    for (int j = 0; j < CR; ++j) {
      float vj = v[j];
      rank += (vj > vi) || (vj == vi && j < i);   // stable: lower index wins ties
    }
    if (rank < CRNEW) outIdx[(size_t)bh * CRNEW + rank] = i;
  }
}

extern "C" void kernel_launch(void* const* d_in, const int* in_sizes, int n_in,
                              void* d_out, int out_size, void* d_ws, size_t ws_size,
                              hipStream_t stream) {
  const float* x    = (const float*)d_in[0];
  const float* wq   = (const float*)d_in[1];
  const float* bq   = (const float*)d_in[2];
  const float* wk   = (const float*)d_in[3];
  const float* bk   = (const float*)d_in[4];
  const float* wv   = (const float*)d_in[5];
  const float* bv   = (const float*)d_in[6];
  const float* wo   = (const float*)d_in[7];
  const float* bo   = (const float*)d_in[8];
  const int*   mask = (const int*)d_in[9];
  const int*   topk = (const int*)d_in[10];

  float* out = (float*)d_out;
  int* newtopk = (int*)((float*)d_out + (size_t)CB * CS * CD);  // tuple: out then indices

  // workspace carve-up (all 256B aligned)
  char* wsp = (char*)d_ws;
  size_t off = 0;
  auto alloc = [&](size_t bytes) {
    void* p = wsp + off;
    off += bytes; off = (off + 255) & ~(size_t)255;
    return p;
  };
  bf16_t* xb   = (bf16_t*)alloc((size_t)CB * CS * CD * 2);
  bf16_t* wqb  = (bf16_t*)alloc((size_t)CD * CD * 2);
  bf16_t* wkb  = (bf16_t*)alloc((size_t)CD * CD * 2);
  bf16_t* wvb  = (bf16_t*)alloc((size_t)CD * CD * 2);
  bf16_t* wob  = (bf16_t*)alloc((size_t)CD * CD * 2);
  bf16_t* Qw   = (bf16_t*)alloc((size_t)CBH * CS * CHD * 2);
  bf16_t* Kw   = (bf16_t*)alloc((size_t)CBH * CS * CHD * 2);
  bf16_t* Vw   = (bf16_t*)alloc((size_t)CBH * CS * CHD * 2);
  bf16_t* Kp   = (bf16_t*)alloc((size_t)CBH * CR * CHD * 2);
  bf16_t* VpT  = (bf16_t*)alloc((size_t)CBH * CHD * CR * 2);
  bf16_t* attnout = (bf16_t*)alloc((size_t)CB * CS * CD * 2);
  float* colmax     = (float*)alloc((size_t)CBH * CR * 4);
  float* colsum     = (float*)alloc((size_t)CBH * CR * 4);
  float* importance = (float*)alloc((size_t)CBH * CR * 4);
  int*   mg         = (int*)  alloc((size_t)CBH * CR * 4);
  float* part1 = (float*)alloc(32ull * CBH * CR * 4);
  float* part2 = (float*)alloc(32ull * CBH * CR * 4);

  // 1) convert inputs to bf16
  int nx = CB * CS * CD, nw = CD * CD;
  k_cvt_bf16<<<(nx + 255) / 256, 256, 0, stream>>>(x,  xb,  nx);
  k_cvt_bf16<<<(nw + 255) / 256, 256, 0, stream>>>(wq, wqb, nw);
  k_cvt_bf16<<<(nw + 255) / 256, 256, 0, stream>>>(wk, wkb, nw);
  k_cvt_bf16<<<(nw + 255) / 256, 256, 0, stream>>>(wv, wvb, nw);
  k_cvt_bf16<<<(nw + 255) / 256, 256, 0, stream>>>(wo, wob, nw);

  // 2) QKV projections (WMMA bf16), written in head layout
  dim3 gg(CB * CS / 128, CD / 64);
  k_gemm<<<gg, 256, 0, stream>>>(xb, wqb, bq, Qw, nullptr, CB * CS, CD, CD, 0);
  k_gemm<<<gg, 256, 0, stream>>>(xb, wkb, bk, Kw, nullptr, CB * CS, CD, CD, 0);
  k_gemm<<<gg, 256, 0, stream>>>(xb, wvb, bv, Vw, nullptr, CB * CS, CD, CD, 0);

  // 3) gather pruned K/V + mask, init colmax
  k_gather<<<(CBH * CR + 255) / 256, 256, 0, stream>>>(Kw, Vw, topk, mask, Kp, VpT, mg, colmax);

  // 4) column-softmax attention (softmax over queries)
  dim3 ga(CS / 64, CBH);
  k_pass1<<<ga, 128, 0, stream>>>(Qw, Kp, mg, colmax);
  k_pass2<<<ga, 128, 0, stream>>>(Qw, Kp, mg, colmax, part1);
  k_reduce32<<<(CBH * CR + 255) / 256, 256, 0, stream>>>(part1, colsum, CBH * CR);
  k_pass3<<<ga, 128, 0, stream>>>(Qw, Kp, VpT, mg, colmax, colsum, attnout, part2);
  k_reduce32<<<(CBH * CR + 255) / 256, 256, 0, stream>>>(part2, importance, CBH * CR);

  // 5) output projection (fp32 result + bias into d_out)
  k_gemm<<<gg, 256, 0, stream>>>(attnout, wob, bo, nullptr, out, CB * CS, CD, CD, 1);

  // 6) new top-k indices
  k_topk<<<CBH, 256, 0, stream>>>(importance, newtopk);
}
